// V_Gem_52123723105098
// MI455X (gfx1250) — matrix operand order repaired and checked by reading, not compiled, not run on hardware
//
#include <hip/hip_runtime.h>
#include <hip/hip_bf16.h>
#include <math.h>

// CDNA5 / gfx1250: wave32, WMMA 16x16x32 f16 -> f32 accumulate.

typedef __attribute__((ext_vector_type(16))) _Float16 v16h;
typedef __attribute__((ext_vector_type(8)))  _Float16 v8h;
typedef __attribute__((ext_vector_type(4)))  _Float16 v4h;
typedef __attribute__((ext_vector_type(8)))  float    v8f;
typedef __attribute__((ext_vector_type(4)))  float    v4f;

#define DIM 64
#define NEG_SLOPE_C 0.2f
#define EPS_C 1e-5f

// ---------------------------------------------------------------------------
// Weight staging: convert f32 64x64 (K-major) weights into WMMA-B register
// order in LDS. B fragment for (kc, nt, lane): b[e] = W[k*64 + col],
// e = 0..15, k = kc*32 + (e>>3)*16 + (lane>=16)*8 + (e&7), col = nt*16 + (lane&15).
// Stored contiguously: lWs[((kc*4+nt)*32 + lane)*16 + e]  -> one v16h load.
// ---------------------------------------------------------------------------
__device__ __forceinline__ void stage_weights_swz(const float* __restrict__ W,
                                                  _Float16* __restrict__ lWs,
                                                  int tid) {
  for (int i = tid; i < 4096; i += 128) {
    const int e  = i & 15;
    const int ln = (i >> 4) & 31;
    const int nt = (i >> 9) & 3;
    const int kc = i >> 11;
    const int col = nt * 16 + (ln & 15);
    const int k   = kc * 32 + (e >> 3) * 16 + (ln >> 4) * 8 + (e & 7);
    lWs[i] = (_Float16)W[k * DIM + col];
  }
}

// One wave computes a 16x64 tile of (A[16x64] @ W[64x64]) into acc[4] (f32).
// lA: wave-private LDS tile, row-major [16][64] f16, 16B aligned.
// lWs: pre-swizzled weights (see above), 32B aligned v16h array.
__device__ __forceinline__ void wave_gemm64(const _Float16* __restrict__ lA,
                                            const v16h* __restrict__ lWs,
                                            int lane, v8f acc[4]) {
  const int rc = lane & 15;   // A row within 16-tile
  const int hi = lane >> 4;   // half-wave selector
  const _Float16* arow = lA + rc * DIM;
#pragma unroll
  for (int kc = 0; kc < 2; ++kc) {
    const int ka = kc * 32 + hi * 8;
    // A fragment = two contiguous 8-half chunks (16B each).
    const v8h alo = *(const v8h*)(arow + ka);
    const v8h ahi = *(const v8h*)(arow + ka + 16);
    const v16h a = __builtin_shufflevector(alo, ahi,
        0, 1, 2, 3, 4, 5, 6, 7, 8, 9, 10, 11, 12, 13, 14, 15);
#pragma unroll
    for (int nt = 0; nt < 4; ++nt) {
      const v16h b = lWs[(kc * 4 + nt) * 32 + lane];
      acc[nt] = __builtin_amdgcn_wmma_f32_16x16x32_f16(
          false, a, false, b, (short)0, acc[nt], false, false);
    }
  }
}

// ---------------- Node pooling: Vp = leaky(leaky(V) @ pAw + pAb) @ pBw + pBb
__global__ __launch_bounds__(128)
void node_pool_kernel(const float* __restrict__ V,
                      const float* __restrict__ pAw, const float* __restrict__ pAb,
                      const float* __restrict__ pBw, const float* __restrict__ pBb,
                      float* __restrict__ Vp, int n_nodes, int n_tiles) {
  __shared__ v16h lWA[256];                     // 8KB, swizzled
  __shared__ v16h lWB[256];                     // 8KB, swizzled
  __shared__ float lbA[DIM], lbB[DIM];
  __shared__ __align__(16) _Float16 bufA[4][16 * DIM];
  __shared__ __align__(16) _Float16 bufX[4][16 * DIM];

  const int tid  = threadIdx.x;
  const int lane = tid & 31;
  const int w    = tid >> 5;

  stage_weights_swz(pAw, (_Float16*)lWA, tid);
  stage_weights_swz(pBw, (_Float16*)lWB, tid);
  if (tid < DIM) { lbA[tid] = pAb[tid]; lbB[tid] = pBb[tid]; }
  __syncthreads();

  const int rc = lane & 15, hi = lane >> 4;

  for (int tile = blockIdx.x; tile < n_tiles; tile += gridDim.x) {
    const int base = tile * 64;
    const int rem  = n_nodes - base;

    // Stage leaky_relu(V) rows wide: float4 load -> v4h store (wave-private).
    for (int c = lane; c < 256; c += 32) {
      const int row  = c >> 4;            // 0..15
      const int col  = (c & 15) * 4;
      const int grow = w * 16 + row;
      v4f v = {};
      if (grow < rem) {
        v = *(const v4f*)(V + (size_t)(base + grow) * DIM + col);
#pragma unroll
        for (int j = 0; j < 4; ++j) v[j] = v[j] > 0.f ? v[j] : NEG_SLOPE_C * v[j];
      }
      *(v4h*)(&bufA[w][row * DIM + col]) = __builtin_convertvector(v, v4h);
    }

    v8f acc[4] = {};
    wave_gemm64(bufA[w], lWA, lane, acc);
#pragma unroll
    for (int nt = 0; nt < 4; ++nt) {
      const int n = nt * 16 + rc;
      const float bias = lbA[n];
#pragma unroll
      for (int r = 0; r < 8; ++r) {
        float v = acc[nt][r] + bias;
        v = v > 0.f ? v : NEG_SLOPE_C * v;
        bufX[w][(hi * 8 + r) * DIM + n] = (_Float16)v;
      }
    }

    v8f acc2[4] = {};
    wave_gemm64(bufX[w], lWB, lane, acc2);
#pragma unroll
    for (int nt = 0; nt < 4; ++nt) {
      const int n = nt * 16 + rc;
      const float bias = lbB[n];
#pragma unroll
      for (int r = 0; r < 8; ++r) {
        const int grow = w * 16 + hi * 8 + r;
        if (grow < rem)
          Vp[(size_t)(base + grow) * DIM + n] = acc2[nt][r] + bias;
      }
    }
  }
}

// ---------------- Edge path: MLP + gate/shift, message, squared-clip, scatter
__global__ __launch_bounds__(128)
void edge_kernel(const float* __restrict__ E,
                 const int* __restrict__ src, const int* __restrict__ dst,
                 const float* __restrict__ w1, const float* __restrict__ b1,
                 const float* __restrict__ w2, const float* __restrict__ b2,
                 const float* __restrict__ Bw, const float* __restrict__ Bb,
                 const float* __restrict__ Cw, const float* __restrict__ Cb,
                 const float* __restrict__ Vp,
                 float* __restrict__ ssum, float* __restrict__ cnt,
                 int n_edges, int n_tiles) {
  __shared__ v16h lW1[256], lW2[256], lGw[256], lSw[256];   // 4 x 8KB swizzled
  __shared__ float lb1[DIM], lb2[DIM], lbB[DIM], lbC[DIM];
  __shared__ __align__(16) _Float16 bufA[4][16 * DIM];
  __shared__ __align__(16) _Float16 bufX[4][16 * DIM];
  __shared__ int lsrc[64], ldst[64];

  const int tid  = threadIdx.x;
  const int lane = tid & 31;
  const int w    = tid >> 5;

  stage_weights_swz(w1, (_Float16*)lW1, tid);
  stage_weights_swz(w2, (_Float16*)lW2, tid);
  stage_weights_swz(Bw, (_Float16*)lGw, tid);
  stage_weights_swz(Cw, (_Float16*)lSw, tid);
  if (tid < DIM) { lb1[tid] = b1[tid]; lb2[tid] = b2[tid]; lbB[tid] = Bb[tid]; lbC[tid] = Cb[tid]; }
  __syncthreads();

  const int rc = lane & 15, hi = lane >> 4;

  for (int tile = blockIdx.x; tile < n_tiles; tile += gridDim.x) {
    const int base = tile * 64;
    const int rem  = n_edges - base;

    if (tid < 64) {
      const bool ok = tid < rem;
      lsrc[tid] = ok ? src[base + tid] : 0;
      ldst[tid] = ok ? dst[base + tid] : 0;
    }
    // Stage E rows wide: float4 load -> v4h store (wave-private).
    for (int c = lane; c < 256; c += 32) {
      const int row  = c >> 4;
      const int col  = (c & 15) * 4;
      const int grow = w * 16 + row;
      v4f v = {};
      if (grow < rem)
        v = *(const v4f*)(E + (size_t)(base + grow) * DIM + col);
      *(v4h*)(&bufA[w][row * DIM + col]) = __builtin_convertvector(v, v4h);
    }
    __syncthreads();  // lsrc/ldst visible to all waves

    // x1 = relu(E @ w1 + b1)
    v8f acc[4] = {};
    wave_gemm64(bufA[w], lW1, lane, acc);
#pragma unroll
    for (int nt = 0; nt < 4; ++nt) {
      const int n = nt * 16 + rc;
      const float bias = lb1[n];
#pragma unroll
      for (int r = 0; r < 8; ++r) {
        float v = fmaxf(acc[nt][r] + bias, 0.f);
        bufX[w][(hi * 8 + r) * DIM + n] = (_Float16)v;
      }
    }

    // x2 = relu(x1 @ w2 + b2)  -> back into bufA
    v8f acc2[4] = {};
    wave_gemm64(bufX[w], lW2, lane, acc2);
#pragma unroll
    for (int nt = 0; nt < 4; ++nt) {
      const int n = nt * 16 + rc;
      const float bias = lb2[n];
#pragma unroll
      for (int r = 0; r < 8; ++r) {
        float v = fmaxf(acc2[nt][r] + bias, 0.f);
        bufA[w][(hi * 8 + r) * DIM + n] = (_Float16)v;
      }
    }

    // gate logits and shift
    v8f accS[4] = {};
    v8f accC[4] = {};
    wave_gemm64(bufA[w], lGw, lane, accS);
    wave_gemm64(bufA[w], lSw, lane, accC);

    // msg = sigmoid(gate)*Vp[src] + shift ; M = clip(msg,eps)^2 ; scatter-add
#pragma unroll
    for (int nt = 0; nt < 4; ++nt) {
      const int n = nt * 16 + rc;
      const float bG = lbB[n];
      const float bS = lbC[n];
#pragma unroll
      for (int r = 0; r < 8; ++r) {
        const int lrow = w * 16 + hi * 8 + r;
        if (lrow < rem) {
          const float sc = 1.f / (1.f + __expf(-(accS[nt][r] + bG)));
          const float sh = accC[nt][r] + bS;
          const float vp = Vp[(size_t)lsrc[lrow] * DIM + n];
          float m = fmaxf(sc * vp + sh, EPS_C);
          atomicAdd(&ssum[(size_t)ldst[lrow] * DIM + n], m * m);
        }
      }
    }
    if (tid < 64 && tid < rem) atomicAdd(&cnt[ldst[tid]], 1.0f);
    __syncthreads();  // protect lsrc/ldst before next tile
  }
}

__global__ void zero_kernel(float* __restrict__ p, long n) {
  long i = (long)blockIdx.x * blockDim.x + threadIdx.x;
  if (i < n) p[i] = 0.f;
}

__global__ void finalize_kernel(const float* __restrict__ ssum,
                                const float* __restrict__ cnt,
                                float* __restrict__ out, int n_nodes) {
  int i = blockIdx.x * blockDim.x + threadIdx.x;
  if (i < n_nodes * DIM) {
    const int node = i >> 6;  // DIM == 64
    const float c = fmaxf(cnt[node], 1.0f);
    out[i] = sqrtf(ssum[i] / c);   // power(mean, 1/ALPHA) with ALPHA=2
  }
}

extern "C" void kernel_launch(void* const* d_in, const int* in_sizes, int n_in,
                              void* d_out, int out_size, void* d_ws, size_t ws_size,
                              hipStream_t stream) {
  const float* V   = (const float*)d_in[0];
  const float* E   = (const float*)d_in[1];
  const int*   src = (const int*)d_in[2];
  const int*   dst = (const int*)d_in[3];
  const float* w1  = (const float*)d_in[4];
  const float* b1  = (const float*)d_in[5];
  const float* w2  = (const float*)d_in[6];
  const float* b2  = (const float*)d_in[7];
  const float* Bw  = (const float*)d_in[8];
  const float* Bb  = (const float*)d_in[9];
  const float* Cw  = (const float*)d_in[10];
  const float* Cb  = (const float*)d_in[11];
  const float* pAw = (const float*)d_in[12];
  const float* pAb = (const float*)d_in[13];
  const float* pBw = (const float*)d_in[14];
  const float* pBb = (const float*)d_in[15];

  const int n_nodes = in_sizes[0] / DIM;
  const int n_edges = in_sizes[2];

  float* Vp   = (float*)d_ws;
  float* ssum = Vp + (size_t)n_nodes * DIM;
  float* cnt  = ssum + (size_t)n_nodes * DIM;  // contiguous with ssum

  {  // zero ssum + cnt (contiguous region)
    long nz = (long)n_nodes * DIM + n_nodes;
    zero_kernel<<<(int)((nz + 255) / 256), 256, 0, stream>>>(ssum, nz);
  }
  {  // node pooling -> Vp
    int n_tiles = (n_nodes + 63) / 64;
    node_pool_kernel<<<n_tiles, 128, 0, stream>>>(V, pAw, pAb, pBw, pBb, Vp,
                                                  n_nodes, n_tiles);
  }
  {  // edge MLP + message + scatter
    int n_tiles = (n_edges + 63) / 64;
    int grid = n_tiles < 65535 ? n_tiles : 65535;
    edge_kernel<<<grid, 128, 0, stream>>>(E, src, dst, w1, b1, w2, b2,
                                          Bw, Bb, Cw, Cb, Vp, ssum, cnt,
                                          n_edges, n_tiles);
  }
  {  // mean + sqrt
    int n = n_nodes * DIM;
    finalize_kernel<<<(n + 255) / 256, 256, 0, stream>>>(ssum, cnt,
                                                         (float*)d_out, n_nodes);
  }
}
